// ProGenAttention_71562745086240
// MI455X (gfx1250) — matrix operand order repaired and checked by reading, not compiled, not run on hardware
//
#include <hip/hip_runtime.h>
#include <hip/hip_bf16.h>

// Problem constants (B, S, E, H, HD, RD, MP) = (2, 2048, 1024, 16, 64, 32, 8)
#define BB  2
#define SS  2048
#define EE  1024
#define HH  16
#define HDD 64
#define RDD 32
#define NQKV 3072   // 3*E
#define MROWS 4096  // B*S

typedef __attribute__((ext_vector_type(16))) __bf16          v16bf;
typedef __attribute__((ext_vector_type(8)))  float           v8f;
typedef __attribute__((ext_vector_type(16))) unsigned short  v16us;

union BFrag { v16us u; v16bf b; uint4 q[2]; };

__device__ __forceinline__ unsigned short f2bf(float f) {
  union { __bf16 b; unsigned short u; } cv;
  cv.b = (__bf16)f;                 // native v_cvt (RNE)
  return cv.u;
}

__device__ __forceinline__ v8f wmma_bf16(v16bf a, v16bf b, v8f c) {
  // D = A(16x32) * B(32x16) + C, fp32 accumulate
  return __builtin_amdgcn_wmma_f32_16x16x32_bf16(false, a, false, b, (short)0, c, false, false);
}

// ---- Fragment loads.  Operands stored row-major as (rows, K), K contiguous.
// A fragment (16x32, MxK): lane r=l&15, elements 0..7 = K(hi*8+0..7), 8..15 = K(16+hi*8+0..7)
// B fragment (32x16, KxN) for A@B^T (operand stored (N,K)): lane r=l&15, elements 0..7 = K(hi*8+0..7), 8..15 = K(16+hi*8+0..7)

__device__ __forceinline__ v16bf load_a_bf(const unsigned short* p, int ld) {
  const int lane = threadIdx.x & 31;
  const unsigned short* row = p + (size_t)(lane & 15) * ld + ((lane >> 4) << 3);
  BFrag f;
  f.q[0] = *(const uint4*)(row);
  f.q[1] = *(const uint4*)(row + 16);
  return f.b;
}

__device__ __forceinline__ v16bf load_b_bf(const unsigned short* p, int ld) {
  const int lane = threadIdx.x & 31;
  const unsigned short* row = p + (size_t)(lane & 15) * ld + ((lane >> 4) << 3);
  BFrag f;
  f.q[0] = *(const uint4*)(row);
  f.q[1] = *(const uint4*)(row + 16);
  return f.b;
}

#define ZERO8 {0.f,0.f,0.f,0.f,0.f,0.f,0.f,0.f}

// ------------------------------------------------------------------
// Kernel 0: one-time f32 -> bf16 conversion (8 elements/thread, b128 in / b128 out)
__global__ void __launch_bounds__(256) cvt_bf16_kernel(
    const float* __restrict__ src, unsigned short* __restrict__ dst, int n8) {
  const int idx = blockIdx.x * 256 + threadIdx.x;
  if (idx >= n8) return;
  const float4* s = (const float4*)src + (size_t)idx * 2;
  float4 a = s[0], b = s[1];
  union { unsigned short u[8]; uint4 q; } p;
  p.u[0] = f2bf(a.x); p.u[1] = f2bf(a.y); p.u[2] = f2bf(a.z); p.u[3] = f2bf(a.w);
  p.u[4] = f2bf(b.x); p.u[5] = f2bf(b.y); p.u[6] = f2bf(b.z); p.u[7] = f2bf(b.w);
  ((uint4*)dst)[idx] = p.q;
}

// ------------------------------------------------------------------
// Kernel 1: qkv = hs @ w_qkv^T (bf16 operands), fused RoPE, scatter to
// Q/K (b,h,s,d) and Vt (b,h,d,s), all bf16.
// Block = 256 threads = 8 waves as 2(m) x 4(n); wave tile 64(M) x 64(N); block tile 128 x 256.
__global__ void __launch_bounds__(256) qkv_rope_kernel(
    const unsigned short* __restrict__ hsb, const unsigned short* __restrict__ wqkvb,
    unsigned short* __restrict__ qbf, unsigned short* __restrict__ kbf,
    unsigned short* __restrict__ vtbf) {
  const int w = threadIdx.x >> 5, lane = threadIdx.x & 31;
  const int nl = lane & 15, hi = lane >> 4;
  const int m0 = blockIdx.y * 128 + (w >> 2) * 64;
  const int n0 = blockIdx.x * 256 + (w & 3) * 64;

  v8f acc[4][4] = {{ZERO8, ZERO8, ZERO8, ZERO8}, {ZERO8, ZERO8, ZERO8, ZERO8},
                   {ZERO8, ZERO8, ZERO8, ZERO8}, {ZERO8, ZERO8, ZERO8, ZERO8}};
  const unsigned short* arow0 = hsb + (size_t)m0 * EE;
  for (int kk = 0; kk < EE; kk += 32) {
    if (kk + 32 < EE) {  // prefetch next k-step rows (global_prefetch_b8, 64 rows each side)
      __builtin_prefetch(arow0 + (size_t)lane * EE + kk + 32, 0, 3);
      __builtin_prefetch(arow0 + (size_t)(32 + lane) * EE + kk + 32, 0, 3);
      __builtin_prefetch(wqkvb + (size_t)(n0 + lane) * EE + kk + 32, 0, 3);
      __builtin_prefetch(wqkvb + (size_t)(n0 + 32 + lane) * EE + kk + 32, 0, 3);
    }
    // all fragment loads first: 8 load clauses, then 16 WMMAs with staged waits
    v16bf afr[4], bfr[4];
    #pragma unroll
    for (int tm = 0; tm < 4; ++tm)
      afr[tm] = load_a_bf(arow0 + (size_t)(tm * 16) * EE + kk, EE);
    #pragma unroll
    for (int tn = 0; tn < 4; ++tn)
      bfr[tn] = load_b_bf(wqkvb + (size_t)(n0 + tn * 16) * EE + kk, EE);
    #pragma unroll
    for (int tn = 0; tn < 4; ++tn)
      #pragma unroll
      for (int tm = 0; tm < 4; ++tm)
        acc[tm][tn] = wmma_bf16(afr[tm], bfr[tn], acc[tm][tn]);
  }

  // Epilogue: qkv col r -> mp = r/384, t = r%384, which = t/128 (0:q 1:v 2:k),
  //           u = t%128, head = mp*2 + u/64, d = u%64.  RoPE on q,k for d < 32.
  #pragma unroll
  for (int tm = 0; tm < 4; ++tm) {
    #pragma unroll
    for (int tn = 0; tn < 4; ++tn) {
      const int rcol = n0 + tn * 16 + nl;
      const int mp = rcol / 384, t = rcol % 384;
      const int which = t >> 7;         // 0:q 1:v 2:k
      const int u = t & 127;
      const int head = mp * 2 + (u >> 6);
      const int d = u & 63;
      #pragma unroll
      for (int i = 0; i < 8; ++i) {
        const int row  = m0 + tm * 16 + hi * 8 + i;
        const int b    = row >> 11;        // / S
        const int spos = row & (SS - 1);   // % S
        float x = acc[tm][tn][i];
        float partner = __shfl_xor(x, 1, 32);  // neighbor feature (d^1), same row
        float val = x;
        if (which != 1 && d < RDD) {
          const int p = d >> 1;
          const float inv = __powf(10000.0f, -(float)p * 0.0625f);  // 10000^(-p/16)
          float sn, cs;
          __sincosf((float)spos * inv, &sn, &cs);
          // even d: x*cos - x[d+1]*sin ; odd d: x*cos + x[d-1]*sin
          val = x * cs + ((d & 1) ? partner : -partner) * sn;
        }
        const unsigned short bv = f2bf(val);
        const size_t bh = (size_t)(b * HH + head);
        if (which == 0)      qbf [(bh * SS + spos) * HDD + d] = bv;
        else if (which == 2) kbf [(bh * SS + spos) * HDD + d] = bv;
        else                 vtbf[(bh * HDD + d) * SS + spos] = bv;
      }
    }
  }
}

// ------------------------------------------------------------------
// Kernel 2: causal flash attention, 64-wide K blocks.
// Block = 128 threads = 4 waves; each wave owns 32 q rows.  grid = (S/128, B*H).
__global__ void __launch_bounds__(128) attn_kernel(
    const unsigned short* __restrict__ qbf, const unsigned short* __restrict__ kbf,
    const unsigned short* __restrict__ vtbf, unsigned short* __restrict__ ctx) {
  __shared__ __align__(16) unsigned short plds[4][32][72];  // per-wave 32x64 P tile, padded rows
  const int w = threadIdx.x >> 5, lane = threadIdx.x & 31;
  const int nl = lane & 15, hi = lane >> 4;
  const int bh = blockIdx.y, b = bh >> 4, h = bh & 15;
  const int q0 = blockIdx.x * 128 + w * 32;

  const unsigned short* Qp = qbf  + (size_t)bh * SS * HDD;
  const unsigned short* Kp = kbf  + (size_t)bh * SS * HDD;
  const unsigned short* Vt = vtbf + (size_t)bh * HDD * SS;  // (HD, S)

  v16bf qf[2][2];
  #pragma unroll
  for (int tm = 0; tm < 2; ++tm)
    #pragma unroll
    for (int ks = 0; ks < 2; ++ks)
      qf[tm][ks] = load_a_bf(Qp + (size_t)(q0 + tm * 16) * HDD + ks * 32, HDD);

  v8f o[2][4] = {{ZERO8, ZERO8, ZERO8, ZERO8}, {ZERO8, ZERO8, ZERO8, ZERO8}};
  float rm[16], rl[16];
  #pragma unroll
  for (int j = 0; j < 16; ++j) { rm[j] = -3.0e38f; rl[j] = 0.0f; }

  for (int k0 = 0; k0 <= q0; k0 += 64) {   // causal: only lower-triangle blocks
    // ---- scores: 32 q rows x 64 k cols; load 4 K-frags per chunk, then 8 WMMAs
    v8f sc[2][4] = {{ZERO8, ZERO8, ZERO8, ZERO8}, {ZERO8, ZERO8, ZERO8, ZERO8}};
    #pragma unroll
    for (int ks = 0; ks < 2; ++ks) {
      v16bf kf[4];
      #pragma unroll
      for (int tn = 0; tn < 4; ++tn)
        kf[tn] = load_b_bf(Kp + (size_t)(k0 + tn * 16) * HDD + ks * 32, HDD);
      #pragma unroll
      for (int tn = 0; tn < 4; ++tn) {
        sc[0][tn] = wmma_bf16(qf[0][ks], kf[tn], sc[0][tn]);
        sc[1][tn] = wmma_bf16(qf[1][ks], kf[tn], sc[1][tn]);
      }
    }

    // ---- scale + causal mask
    #pragma unroll
    for (int tm = 0; tm < 2; ++tm)
      #pragma unroll
      for (int tn = 0; tn < 4; ++tn)
        #pragma unroll
        for (int i = 0; i < 8; ++i) {
          const int qrow = q0 + tm * 16 + hi * 8 + i;
          const int kcol = k0 + tn * 16 + nl;
          const float s = sc[tm][tn][i] * 0.125f;   // 1/sqrt(64)
          sc[tm][tn][i] = (kcol > qrow) ? -1.0e9f : s;
        }

    // ---- online softmax per row (one pass per 64 columns)
    #pragma unroll
    for (int tm = 0; tm < 2; ++tm)
      #pragma unroll
      for (int i = 0; i < 8; ++i) {
        const int j = tm * 8 + i;
        float mx = fmaxf(fmaxf(sc[tm][0][i], sc[tm][1][i]),
                         fmaxf(sc[tm][2][i], sc[tm][3][i]));
        #pragma unroll
        for (int off = 1; off < 16; off <<= 1) mx = fmaxf(mx, __shfl_xor(mx, off, 32));
        const float mn = fmaxf(rm[j], mx);
        const float corr = __expf(rm[j] - mn);
        const float e0 = __expf(sc[tm][0][i] - mn);
        const float e1 = __expf(sc[tm][1][i] - mn);
        const float e2 = __expf(sc[tm][2][i] - mn);
        const float e3 = __expf(sc[tm][3][i] - mn);
        float rs = (e0 + e1) + (e2 + e3);
        #pragma unroll
        for (int off = 1; off < 16; off <<= 1) rs += __shfl_xor(rs, off, 32);
        rl[j] = rl[j] * corr + rs;
        rm[j] = mn;
        #pragma unroll
        for (int t = 0; t < 4; ++t) o[tm][t][i] *= corr;
        const int prow = tm * 16 + hi * 8 + i;
        plds[w][prow][nl]      = f2bf(e0);
        plds[w][prow][16 + nl] = f2bf(e1);
        plds[w][prow][32 + nl] = f2bf(e2);
        plds[w][prow][48 + nl] = f2bf(e3);
      }

    // ---- O += P @ V over two 32-wide K chunks
    #pragma unroll
    for (int kc = 0; kc < 2; ++kc) {
      v16bf vf[4];
      #pragma unroll
      for (int t = 0; t < 4; ++t)
        vf[t] = load_b_bf(Vt + (size_t)(t * 16) * SS + k0 + kc * 32, SS);
      #pragma unroll
      for (int tm = 0; tm < 2; ++tm) {
        v16bf pa = load_a_bf(&plds[w][tm * 16][kc * 32], 72);
        #pragma unroll
        for (int t = 0; t < 4; ++t)
          o[tm][t] = wmma_bf16(pa, vf[t], o[tm][t]);
      }
    }
  }

  // normalize and store context bf16 at (b*S + q, h*64 + d)
  #pragma unroll
  for (int tm = 0; tm < 2; ++tm)
    #pragma unroll
    for (int i = 0; i < 8; ++i) {
      const int qrow = q0 + tm * 16 + hi * 8 + i;
      const float inv = 1.0f / rl[tm * 8 + i];
      #pragma unroll
      for (int t = 0; t < 4; ++t) {
        const int d = t * 16 + nl;
        ctx[((size_t)b * SS + qrow) * EE + h * HDD + d] = f2bf(o[tm][t][i] * inv);
      }
    }
}

// ------------------------------------------------------------------
// Kernel 3: out = ctx @ w_out^T (bf16 operands, fp32 result).  Wave tile 64x64.
__global__ void __launch_bounds__(256) outproj_kernel(
    const unsigned short* __restrict__ ctx, const unsigned short* __restrict__ woutb,
    float* __restrict__ out) {
  const int w = threadIdx.x >> 5, lane = threadIdx.x & 31;
  const int nl = lane & 15, hi = lane >> 4;
  const int m0 = blockIdx.y * 128 + (w >> 2) * 64;
  const int n0 = blockIdx.x * 256 + (w & 3) * 64;

  v8f acc[4][4] = {{ZERO8, ZERO8, ZERO8, ZERO8}, {ZERO8, ZERO8, ZERO8, ZERO8},
                   {ZERO8, ZERO8, ZERO8, ZERO8}, {ZERO8, ZERO8, ZERO8, ZERO8}};
  const unsigned short* arow0 = ctx + (size_t)m0 * EE;
  for (int kk = 0; kk < EE; kk += 32) {
    if (kk + 32 < EE) {
      __builtin_prefetch(arow0 + (size_t)lane * EE + kk + 32, 0, 3);
      __builtin_prefetch(arow0 + (size_t)(32 + lane) * EE + kk + 32, 0, 3);
      __builtin_prefetch(woutb + (size_t)(n0 + lane) * EE + kk + 32, 0, 3);
      __builtin_prefetch(woutb + (size_t)(n0 + 32 + lane) * EE + kk + 32, 0, 3);
    }
    v16bf afr[4], bfr[4];
    #pragma unroll
    for (int tm = 0; tm < 4; ++tm)
      afr[tm] = load_a_bf(arow0 + (size_t)(tm * 16) * EE + kk, EE);
    #pragma unroll
    for (int tn = 0; tn < 4; ++tn)
      bfr[tn] = load_b_bf(woutb + (size_t)(n0 + tn * 16) * EE + kk, EE);
    #pragma unroll
    for (int tn = 0; tn < 4; ++tn)
      #pragma unroll
      for (int tm = 0; tm < 4; ++tm)
        acc[tm][tn] = wmma_bf16(afr[tm], bfr[tn], acc[tm][tn]);
  }
  #pragma unroll
  for (int tm = 0; tm < 4; ++tm)
    #pragma unroll
    for (int tn = 0; tn < 4; ++tn)
      #pragma unroll
      for (int i = 0; i < 8; ++i)
        out[(size_t)(m0 + tm * 16 + hi * 8 + i) * EE + n0 + tn * 16 + nl] = acc[tm][tn][i];
}

// ------------------------------------------------------------------
extern "C" void kernel_launch(void* const* d_in, const int* in_sizes, int n_in,
                              void* d_out, int out_size, void* d_ws, size_t ws_size,
                              hipStream_t stream) {
  const float* hs   = (const float*)d_in[0];   // (B,S,E)
  const float* wqkv = (const float*)d_in[1];   // (3E,E)
  const float* wout = (const float*)d_in[2];   // (E,E)
  float* out = (float*)d_out;                  // (B,S,E)

  // workspace (bf16 shorts): Q,K (b,h,s,d) | Vt (b,h,d,s) | ctx | hs_bf | wqkv_bf | wout_bf
  const size_t qkvN = (size_t)BB * HH * SS * HDD;        // 4,194,304
  const size_t nHS  = (size_t)MROWS * EE;                // 4,194,304
  const size_t nWQ  = (size_t)NQKV * EE;                 // 3,145,728
  const size_t nWO  = (size_t)EE * EE;                   // 1,048,576
  unsigned short* qbf   = (unsigned short*)d_ws;
  unsigned short* kbf   = qbf + qkvN;
  unsigned short* vtbf  = kbf + qkvN;
  unsigned short* ctx   = vtbf + qkvN;
  unsigned short* hsb   = ctx + nHS;
  unsigned short* wqkvb = hsb + nHS;
  unsigned short* woutb = wqkvb + nWQ;

  (void)in_sizes; (void)n_in; (void)out_size; (void)ws_size;

  // one-time f32 -> bf16 conversions
  cvt_bf16_kernel<<<dim3((nHS / 8 + 255) / 256), dim3(256), 0, stream>>>(hs, hsb, (int)(nHS / 8));
  cvt_bf16_kernel<<<dim3((nWQ / 8 + 255) / 256), dim3(256), 0, stream>>>(wqkv, wqkvb, (int)(nWQ / 8));
  cvt_bf16_kernel<<<dim3((nWO / 8 + 255) / 256), dim3(256), 0, stream>>>(wout, woutb, (int)(nWO / 8));

  dim3 b1(256), g1(NQKV / 256, MROWS / 128);       // (12, 32)
  qkv_rope_kernel<<<g1, b1, 0, stream>>>(hsb, wqkvb, qbf, kbf, vtbf);

  dim3 b2(128), g2(SS / 128, BB * HH);             // (16, 32)
  attn_kernel<<<g2, b2, 0, stream>>>(qbf, kbf, vtbf, ctx);

  dim3 b3(256), g3(EE / 256, MROWS / 128);         // (4, 32)
  outproj_kernel<<<g3, b3, 0, stream>>>(ctx, woutb, out);
}
